// PointTransformer_model_16750372454757
// MI455X (gfx1250) — compile-verified, matrix operand order checked by reading, not run on hardware
//
#include <hip/hip_runtime.h>
#include <hip/hip_bf16.h>

typedef __attribute__((ext_vector_type(16))) _Float16 v16h;
typedef __attribute__((ext_vector_type(8)))  _Float16 h8v;
typedef __attribute__((ext_vector_type(8)))  float    v8f;

// ---------------------------------------------------------------------------
// WMMA GEMM:  C[M,N] = A[M,Kp] * Wh[N,Kp]^T (+ bias[N])
// A fp32 (zero-padded to Kp, Kp % 32 == 0), Wh fp16 pre-converted/padded.
// M, N multiples of 16 (true for every call in this network) -> no guards.
// Fragment layout per ISA: lane holds K runs [k0+half*8, +8) and
// [k0+16+half*8, +8)  => 4x float4 (A) + 2x 16B f16 (B) aligned loads,
// no LDS, no barriers, one v_wmma per 32-wide K chunk.
// ---------------------------------------------------------------------------
__global__ void gemm_wmma_kernel(const float* __restrict__ A,
                                 const _Float16* __restrict__ Bh,
                                 const float* __restrict__ bias,
                                 float* __restrict__ C,
                                 int N, int Kp) {
    const int lane = threadIdx.x;          // 0..31
    const int r    = lane & 15;
    const int half = lane >> 4;
    const int arow = blockIdx.x * 16 + r;  // M row for A fragment
    const int brow = blockIdx.y * 16 + r;  // N row for B fragment
    const float*    Ar = A  + (long long)arow * Kp + half * 8;
    const _Float16* Br = Bh + (long long)brow * Kp + half * 8;
    v8f acc = {};
    for (int k0 = 0; k0 < Kp; k0 += 32) {
        float4 a0 = *(const float4*)(Ar + k0);
        float4 a1 = *(const float4*)(Ar + k0 + 4);
        float4 a2 = *(const float4*)(Ar + k0 + 16);
        float4 a3 = *(const float4*)(Ar + k0 + 20);
        h8v    b0 = *(const h8v*)(Br + k0);
        h8v    b1 = *(const h8v*)(Br + k0 + 16);
        v16h af, bf;
        af[0]  = (_Float16)a0.x; af[1]  = (_Float16)a0.y;
        af[2]  = (_Float16)a0.z; af[3]  = (_Float16)a0.w;
        af[4]  = (_Float16)a1.x; af[5]  = (_Float16)a1.y;
        af[6]  = (_Float16)a1.z; af[7]  = (_Float16)a1.w;
        af[8]  = (_Float16)a2.x; af[9]  = (_Float16)a2.y;
        af[10] = (_Float16)a2.z; af[11] = (_Float16)a2.w;
        af[12] = (_Float16)a3.x; af[13] = (_Float16)a3.y;
        af[14] = (_Float16)a3.z; af[15] = (_Float16)a3.w;
#pragma unroll
        for (int i = 0; i < 8; ++i) { bf[i] = b0[i]; bf[8 + i] = b1[i]; }
        acc = __builtin_amdgcn_wmma_f32_16x16x32_f16(
            /*neg_a=*/false, af, /*neg_b=*/false, bf,
            /*c_mod=*/(short)0, acc, /*reuse_a=*/false, /*reuse_b=*/false);
    }
    const int nc = blockIdx.y * 16 + r;
    float badd = bias ? bias[nc] : 0.f;
    float* Cr = C + (long long)(blockIdx.x * 16) * N + nc;
#pragma unroll
    for (int r8 = 0; r8 < 8; ++r8)
        Cr[(long long)(r8 + 8 * half) * N] = acc[r8] + badd;   // D: VGPR r -> row r+8*half
}

// One-shot weight conversion: W[N,K] fp32 -> Wh[N,Kp] fp16, zero-padded.
__global__ void cvtw_kernel(const float* __restrict__ W, _Float16* __restrict__ Wh,
                            int K, int Kp, int total) {
    int tid = blockIdx.x * 256 + threadIdx.x;
    if (tid >= total) return;
    int k = tid % Kp, n = tid / Kp;
    Wh[tid] = (k < K) ? (_Float16)W[(long long)n * K + k] : (_Float16)0.f;
}

// ---------------------------------------------------------------------------
// Elementwise / reduction helpers
// ---------------------------------------------------------------------------
// concat(xyz, feats) padded to 32 columns (K=6 -> Kp=32)
__global__ void concat6_kernel(const float* __restrict__ p, const float* __restrict__ x,
                               float* __restrict__ out, int n) {
    int tid = blockIdx.x * 256 + threadIdx.x;
    if (tid >= n * 32) return;
    int c = tid % 32, rr = tid / 32;
    float v = 0.f;
    if (c < 3)      v = p[3 * rr + c];
    else if (c < 6) v = x[3 * rr + (c - 3)];
    out[tid] = v;
}

__global__ void meanvar_kernel(const float* __restrict__ Y, int R, int C,
                               float* __restrict__ mu, float* __restrict__ rs) {
    __shared__ float ss[256], sq[256];
    int c = blockIdx.x, t = threadIdx.x;
    float s = 0.f, q = 0.f;
    for (int rr = t; rr < R; rr += 256) {
        float v = Y[(long long)rr * C + c];
        s += v; q += v * v;
    }
    ss[t] = s; sq[t] = q;
    __syncthreads();
    for (int k = 128; k > 0; k >>= 1) {
        if (t < k) { ss[t] += ss[t + k]; sq[t] += sq[t + k]; }
        __syncthreads();
    }
    if (t == 0) {
        float mean = ss[0] / (float)R;
        float var  = sq[0] / (float)R - mean * mean;
        if (var < 0.f) var = 0.f;
        mu[c] = mean;
        rs[c] = rsqrtf(var + 1e-5f);
    }
}

__global__ void bnrelu_kernel(const float* __restrict__ Y, float* __restrict__ O,
                              const float* __restrict__ mu, const float* __restrict__ rs,
                              const float* __restrict__ g, const float* __restrict__ b,
                              int total, int C) {
    int tid = blockIdx.x * 256 + threadIdx.x;
    if (tid >= total) return;
    int c = tid % C;
    float v = g[c] * (Y[tid] - mu[c]) * rs[c] + b[c];
    O[tid] = v > 0.f ? v : 0.f;
}

__global__ void maxpool_kernel(const float* __restrict__ H, float* __restrict__ O,
                               int m, int ns, int C) {
    int tid = blockIdx.x * 256 + threadIdx.x;
    if (tid >= m * C) return;
    int c = tid % C, j = tid / C;
    float mx = -3.4e38f;
    for (int s = 0; s < ns; ++s)
        mx = fmaxf(mx, H[((long long)j * ns + s) * C + c]);
    O[tid] = mx;
}

// ---------------------------------------------------------------------------
// Furthest-point sampling: one persistent workgroup, LDS argmax per iteration.
// ---------------------------------------------------------------------------
__global__ void fps_kernel(const float* __restrict__ p, int n, int m,
                           int* __restrict__ idxs, float* __restrict__ dist) {
    __shared__ float sval[1024];
    __shared__ int   sidx[1024];
    __shared__ float lastp[3];
    int t = threadIdx.x;
    for (int i = t; i < n; i += 1024) dist[i] = 1e10f;
    if (t == 0) { idxs[0] = 0; lastp[0] = p[0]; lastp[1] = p[1]; lastp[2] = p[2]; }
    __syncthreads();
    for (int it = 1; it < m; ++it) {
        float lx = lastp[0], ly = lastp[1], lz = lastp[2];
        float best = -1.f; int bi = 0;
        for (int i = t; i < n; i += 1024) {
            float dx = p[3 * i] - lx, dy = p[3 * i + 1] - ly, dz = p[3 * i + 2] - lz;
            float d = dx * dx + dy * dy + dz * dz;
            float dd = fminf(dist[i], d);
            dist[i] = dd;
            if (dd > best) { best = dd; bi = i; }
        }
        sval[t] = best; sidx[t] = bi;
        __syncthreads();
        for (int s = 512; s > 0; s >>= 1) {
            if (t < s) {
                bool take = (sval[t + s] > sval[t]) ||
                            (sval[t + s] == sval[t] && sidx[t + s] < sidx[t]);
                if (take) { sval[t] = sval[t + s]; sidx[t] = sidx[t + s]; }
            }
            __syncthreads();
        }
        if (t == 0) {
            int ci = sidx[0];
            idxs[it] = ci;
            lastp[0] = p[3 * ci]; lastp[1] = p[3 * ci + 1]; lastp[2] = p[3 * ci + 2];
        }
        __syncthreads();
    }
}

__global__ void gatherp_kernel(const float* __restrict__ p, const int* __restrict__ idx,
                               float* __restrict__ out, int m) {
    int tid = blockIdx.x * 256 + threadIdx.x;
    if (tid >= 3 * m) return;
    int j = tid / 3, c = tid % 3;
    out[tid] = p[3 * idx[j] + c];
}

// ---------------------------------------------------------------------------
// KNN: one thread per query, register insertion-sorted top-K + prefetch.
// ---------------------------------------------------------------------------
template <int K>
__global__ void knn_kernel(const float* __restrict__ q, int m,
                           const float* __restrict__ p, int n,
                           int* __restrict__ outIdx) {
    int j = blockIdx.x * 64 + threadIdx.x;
    if (j >= m) return;
    float qx = q[3 * j], qy = q[3 * j + 1], qz = q[3 * j + 2];
    float bd[K]; int bi[K];
#pragma unroll
    for (int s = 0; s < K; ++s) { bd[s] = 3.4e38f; bi[s] = 0; }
    for (int i = 0; i < n; ++i) {
        if ((i & 31) == 0 && i + 64 < n)
            __builtin_prefetch(&p[3 * (i + 64)], 0, 1);
        float dx = p[3 * i] - qx, dy = p[3 * i + 1] - qy, dz = p[3 * i + 2] - qz;
        float d = dx * dx + dy * dy + dz * dz;
        if (d < bd[K - 1]) {
            bd[K - 1] = d; bi[K - 1] = i;
#pragma unroll
            for (int s = K - 1; s > 0; --s) {
                if (bd[s] < bd[s - 1]) {
                    float tf = bd[s]; bd[s] = bd[s - 1]; bd[s - 1] = tf;
                    int ti = bi[s]; bi[s] = bi[s - 1]; bi[s - 1] = ti;
                }
            }
        }
    }
#pragma unroll
    for (int s = 0; s < K; ++s) outIdx[j * K + s] = bi[s];
}

__global__ void knn3w_kernel(const float* __restrict__ q, int m,
                             const float* __restrict__ p, int n,
                             int* __restrict__ iidx, float* __restrict__ iw) {
    int j = blockIdx.x * 64 + threadIdx.x;
    if (j >= m) return;
    float qx = q[3 * j], qy = q[3 * j + 1], qz = q[3 * j + 2];
    float bd[3] = {3.4e38f, 3.4e38f, 3.4e38f};
    int   bi[3] = {0, 0, 0};
    for (int i = 0; i < n; ++i) {
        if ((i & 31) == 0 && i + 64 < n)
            __builtin_prefetch(&p[3 * (i + 64)], 0, 1);
        float dx = p[3 * i] - qx, dy = p[3 * i + 1] - qy, dz = p[3 * i + 2] - qz;
        float d = dx * dx + dy * dy + dz * dz;
        if (d < bd[2]) {
            bd[2] = d; bi[2] = i;
            if (bd[2] < bd[1]) { float tf = bd[2]; bd[2] = bd[1]; bd[1] = tf; int ti = bi[2]; bi[2] = bi[1]; bi[1] = ti; }
            if (bd[1] < bd[0]) { float tf = bd[1]; bd[1] = bd[0]; bd[0] = tf; int ti = bi[1]; bi[1] = bi[0]; bi[0] = ti; }
        }
    }
    float w0 = 1.f / (bd[0] + 1e-8f);
    float w1 = 1.f / (bd[1] + 1e-8f);
    float w2 = 1.f / (bd[2] + 1e-8f);
    float s = w0 + w1 + w2;
    iw[3 * j] = w0 / s; iw[3 * j + 1] = w1 / s; iw[3 * j + 2] = w2 / s;
    iidx[3 * j] = bi[0]; iidx[3 * j + 1] = bi[1]; iidx[3 * j + 2] = bi[2];
}

// grouped = [p[nidx]-np | x[nidx] | 0-pad]  written with padded row stride Kp
__global__ void group_kernel(const int* __restrict__ nidx, const float* __restrict__ p,
                             const float* __restrict__ np, const float* __restrict__ x,
                             float* __restrict__ G, int m, int ns, int Cx, int Kp) {
    int tid = blockIdx.x * 256 + threadIdx.x;
    int total = m * ns * Kp;
    if (tid >= total) return;
    int c = tid % Kp;
    int row = tid / Kp;
    int s = row % ns;
    int j = row / ns;
    int pi = nidx[j * ns + s];
    float v = 0.f;
    if (c < 3)           v = p[3 * pi + c] - np[3 * j + c];
    else if (c < 3 + Cx) v = x[(long long)pi * Cx + (c - 3)];
    G[(long long)row * Kp + c] = v;
}

__global__ void colmean_kernel(const float* __restrict__ X, float* __restrict__ out,
                               int R, int C) {
    int c = blockIdx.x * 256 + threadIdx.x;
    if (c >= C) return;
    float s = 0.f;
    for (int rr = 0; rr < R; ++rr) s += X[(long long)rr * C + c];
    out[c] = s / (float)R;
}

__global__ void vecmat_relu_kernel(const float* __restrict__ v, const float* __restrict__ W,
                                   const float* __restrict__ b, float* __restrict__ out,
                                   int In, int Out) {
    int o = blockIdx.x * 256 + threadIdx.x;
    if (o >= Out) return;
    float s = b[o];
    for (int i = 0; i < In; ++i) s += v[i] * W[(long long)o * In + i];
    out[o] = s > 0.f ? s : 0.f;
}

__global__ void xcat_kernel(const float* __restrict__ x5, const float* __restrict__ gf,
                            float* __restrict__ out, int R) {
    int tid = blockIdx.x * 256 + threadIdx.x;
    if (tid >= R * 1024) return;
    int c = tid % 1024, rr = tid / 1024;
    out[tid] = (c < 512) ? x5[rr * 512 + c] : gf[c - 512];
}

__global__ void interpadd_kernel(const float* __restrict__ a, const float* __restrict__ bb,
                                 const int* __restrict__ iidx, const float* __restrict__ iw,
                                 float* __restrict__ out, int m, int C) {
    int tid = blockIdx.x * 256 + threadIdx.x;
    if (tid >= m * C) return;
    int c = tid % C, j = tid / C;
    float s = a[tid];
#pragma unroll
    for (int t = 0; t < 3; ++t)
        s += iw[3 * j + t] * bb[(long long)iidx[3 * j + t] * C + c];
    out[tid] = s;
}

// ---------------------------------------------------------------------------
// Host pipeline
// ---------------------------------------------------------------------------
extern "C" void kernel_launch(void* const* d_in, const int* in_sizes, int n_in,
                              void* d_out, int out_size, void* d_ws, size_t ws_size,
                              hipStream_t stream) {
    (void)in_sizes; (void)n_in; (void)out_size; (void)ws_size;
    auto F = [&](int i) { return (const float*)d_in[i]; };
    const float* p0 = F(0);
    const float* x0 = F(1);
    const float *encW[5], *encG[5], *encB[5];
    for (int l = 0; l < 5; ++l) { encW[l] = F(2 + 3 * l); encG[l] = F(3 + 3 * l); encB[l] = F(4 + 3 * l); }
    const float* d5W1 = F(17); const float* d5b1 = F(18);
    const float* d5g1 = F(19); const float* d5be1 = F(20);
    const float* d5W2 = F(21); const float* d5b2 = F(22);

    // Workspace bump allocator; all regions multiples of 4 floats -> 16B aligned.
    float* WSP = (float*)d_ws;
    size_t off = 0;
    auto alloc  = [&](size_t nf) { float* r = WSP + off; off += nf; return r; };
    auto halloc = [&](size_t nh) { _Float16* r = (_Float16*)(WSP + off); off += nh / 2; return r; };
    float* X6 = alloc(16384 * 32);           // padded enc0 input (Kp=32)
    float* P2 = alloc(4096 * 3); float* P3 = alloc(1024 * 3);
    float* P4 = alloc(256 * 3);  float* P5 = alloc(64 * 3);
    float* X1 = alloc(16384 * 32); float* X2 = alloc(4096 * 64);
    float* X3 = alloc(1024 * 128); float* X4 = alloc(256 * 256);
    float* X5 = alloc(64 * 512);
    int*   FIDX = (int*)alloc(4096);
    int*   KNNI = (int*)alloc(65536);
    int*   IIDX = (int*)alloc(16384 * 3);
    float* IW   = alloc(16384 * 3);
    float* DIST = alloc(16384);
    float* MU = alloc(512); float* RS = alloc(512);
    float* GMEAN = alloc(512); float* GFEAT = alloc(512);
    // fp16 padded weights
    const int encKp[5] = {32, 64, 96, 160, 288};     // pad(6,35,67,131,259)
    _Float16* WHenc[5];
    for (int l = 0; l < 5; ++l) WHenc[l] = halloc((size_t)((l == 0) ? 32 : 64 << (l - 1)) * encKp[l] * ((l == 0) ? 1 : 1));
    // (sizes: 32*32, 64*64, 128*96, 256*160, 512*288)
    // recompute properly:
    // -- handled below with explicit sizes
    _Float16* WHd5W1 = halloc(512 * 1024);
    _Float16* WHdec1[4]; _Float16* WHdec2[4];
    const int decC[4]  = {256, 128, 64, 32};
    const int decK2[4] = {512, 256, 128, 64};
    for (int i = 0; i < 4; ++i) { WHdec1[i] = halloc((size_t)decC[i] * decC[i]); }
    for (int i = 0; i < 4; ++i) { WHdec2[i] = halloc((size_t)decC[i] * decK2[i]); }
    float* BUFA = alloc(4194304);
    float* BUFB = alloc(4194304);

    float* OUT = (float*)d_out;
    float* O1 = OUT;               // x1d [16384,32]
    float* O2 = O1 + 16384 * 32;   // x2d [4096,64]
    float* O3 = O2 + 4096 * 64;    // x3d [1024,128]
    float* O4 = O3 + 1024 * 128;   // x4d [256,256]
    float* O5 = O4 + 256 * 256;    // x5d [64,512]

    auto cvtw = [&](const float* W, _Float16* Wh, int N, int K, int Kp) {
        int total = N * Kp;
        cvtw_kernel<<<dim3((total + 255) / 256), dim3(256), 0, stream>>>(W, Wh, K, Kp, total);
    };
    auto gemm = [&](const float* A, const _Float16* Bh, const float* bias, float* C,
                    int M, int N, int Kp) {
        gemm_wmma_kernel<<<dim3(M / 16, N / 16), dim3(32), 0, stream>>>(A, Bh, bias, C, N, Kp);
    };
    auto meanvar = [&](const float* Y, int R, int C) {
        meanvar_kernel<<<dim3(C), dim3(256), 0, stream>>>(Y, R, C, MU, RS);
    };
    auto bnrelu = [&](const float* Y, float* O, const float* g, const float* b,
                      int R, int C) {
        int total = R * C;
        bnrelu_kernel<<<dim3((total + 255) / 256), dim3(256), 0, stream>>>(Y, O, MU, RS, g, b, total, C);
    };

    // ---- one-shot weight conversions (fp32 -> padded fp16) ----
    const int encN[5] = {32, 64, 128, 256, 512};
    const int encK[5] = {6, 35, 67, 131, 259};
    for (int l = 0; l < 5; ++l) cvtw(encW[l], WHenc[l], encN[l], encK[l], encKp[l]);
    cvtw(d5W1, WHd5W1, 512, 1024, 1024);
    const int decBase[4] = {23, 31, 39, 47};
    for (int i = 0; i < 4; ++i) {
        cvtw(F(decBase[i]),     WHdec1[i], decC[i], decC[i],  decC[i]);
        cvtw(F(decBase[i] + 4), WHdec2[i], decC[i], decK2[i], decK2[i]);
    }

    // ---- enc0: Linear(6->32) + BN + ReLU ----
    concat6_kernel<<<dim3((16384 * 32 + 255) / 256), dim3(256), 0, stream>>>(p0, x0, X6, 16384);
    gemm(X6, WHenc[0], nullptr, BUFA, 16384, 32, 32);
    meanvar(BUFA, 16384, 32);
    bnrelu(BUFA, X1, encG[0], encB[0], 16384, 32);

    // ---- encoder levels 1..4 ----
    struct Lv { int n, m, Cx, Cout; const float* pin; float* pout; const float* xin; float* xout; int wi; };
    Lv lv[4] = {
        {16384, 4096, 32,  64,  p0, P2, X1, X2, 1},
        {4096,  1024, 64,  128, P2, P3, X2, X3, 2},
        {1024,  256,  128, 256, P3, P4, X3, X4, 3},
        {256,   64,   256, 512, P4, P5, X4, X5, 4}};
    for (int li = 0; li < 4; ++li) {
        Lv& L = lv[li];
        fps_kernel<<<dim3(1), dim3(1024), 0, stream>>>(L.pin, L.n, L.m, FIDX, DIST);
        gatherp_kernel<<<dim3((3 * L.m + 255) / 256), dim3(256), 0, stream>>>(L.pin, FIDX, L.pout, L.m);
        knn_kernel<16><<<dim3((L.m + 63) / 64), dim3(64), 0, stream>>>(L.pout, L.m, L.pin, L.n, KNNI);
        int Kp = encKp[L.wi];
        int rows = L.m * 16;
        int gt = rows * Kp;
        group_kernel<<<dim3((gt + 255) / 256), dim3(256), 0, stream>>>(KNNI, L.pin, L.pout, L.xin, BUFA, L.m, 16, L.Cx, Kp);
        gemm(BUFA, WHenc[L.wi], nullptr, BUFB, rows, L.Cout, Kp);
        meanvar(BUFB, rows, L.Cout);
        bnrelu(BUFB, BUFB, encG[L.wi], encB[L.wi], rows, L.Cout);
        maxpool_kernel<<<dim3((L.m * L.Cout + 255) / 256), dim3(256), 0, stream>>>(BUFB, L.xout, L.m, 16, L.Cout);
    }

    // ---- dec5 head ----
    colmean_kernel<<<dim3(2), dim3(256), 0, stream>>>(X5, GMEAN, 64, 512);
    vecmat_relu_kernel<<<dim3(2), dim3(256), 0, stream>>>(GMEAN, d5W2, d5b2, GFEAT, 512, 512);
    xcat_kernel<<<dim3((64 * 1024 + 255) / 256), dim3(256), 0, stream>>>(X5, GFEAT, BUFA, 64);
    gemm(BUFA, WHd5W1, d5b1, BUFB, 64, 512, 1024);
    meanvar(BUFB, 64, 512);
    bnrelu(BUFB, O5, d5g1, d5be1, 64, 512);

    // ---- decoder up-stages: a + interp3(bb) ----
    struct Dv { int n1, n2, C, K2; const float* p1; const float* p2; const float* x1; const float* x2; float* o; int base; };
    Dv dv[4] = {
        {256,   64,   256, 512, P4, P5, X4, O5, O4, 23},
        {1024,  256,  128, 256, P3, P4, X3, O4, O3, 31},
        {4096,  1024, 64,  128, P2, P3, X2, O3, O2, 39},
        {16384, 4096, 32,  64,  p0, P2, X1, O2, O1, 47}};
    for (int di = 0; di < 4; ++di) {
        Dv& D = dv[di];
        const float* b1  = F(D.base + 1);
        const float* g1  = F(D.base + 2); const float* be1 = F(D.base + 3);
        const float* b2  = F(D.base + 5);
        const float* g2  = F(D.base + 6); const float* be2 = F(D.base + 7);
        gemm(D.x1, WHdec1[di], b1, BUFA, D.n1, D.C, D.C);
        meanvar(BUFA, D.n1, D.C);
        bnrelu(BUFA, BUFA, g1, be1, D.n1, D.C);
        gemm(D.x2, WHdec2[di], b2, BUFB, D.n2, D.C, D.K2);
        meanvar(BUFB, D.n2, D.C);
        bnrelu(BUFB, BUFB, g2, be2, D.n2, D.C);
        knn3w_kernel<<<dim3((D.n1 + 63) / 64), dim3(64), 0, stream>>>(D.p1, D.n1, D.p2, D.n2, IIDX, IW);
        interpadd_kernel<<<dim3((D.n1 * D.C + 255) / 256), dim3(256), 0, stream>>>(BUFA, BUFB, IIDX, IW, D.o, D.n1, D.C);
    }
}
